// LocalScaledDotProdAttV1_49022756716709
// MI455X (gfx1250) — matrix operand order repaired and checked by reading, not compiled, not run on hardware
//
#include <hip/hip_runtime.h>
#include <hip/hip_bf16.h>
#include <math.h>
#include <stdint.h>

// ---------------------------------------------------------------------------
// LocalScaledDotProdAtt for MI455X (gfx1250, wave32, WMMA 16x16x32 f16)
//   B=8 T=4096 F=512 H=8 DK=DV=64 CTX=64 -> nb=64, cq=ck=64, s1=s2=32
// Pipeline:
//   K0: convert Wq/Wk/Wv/Wo f32->f16 once (kills per-tile weight cvt)
//   K1: fused QKV projection, A-resident: 16 A-frags (128 VGPRs) loaded once,
//       reused across all 32 column tiles -> activations read once from HBM
//   K2: local attention; V staged to LDS with GLOBAL_LOAD_ASYNC_TO_LDS_B128
//       (overlapped with QK^T WMMAs), softmax + LDS transpose, P@V WMMAs
//   K3: output projection, A-resident, f32 out
// Workspace: 4 * 32768*512 f16 + 4 * 512*512 f16 = 136,314,880 bytes
// ---------------------------------------------------------------------------

typedef _Float16 v16h __attribute__((ext_vector_type(16)));
typedef _Float16 v8h  __attribute__((ext_vector_type(8)));
typedef _Float16 v4h  __attribute__((ext_vector_type(4)));
typedef float    v8f  __attribute__((ext_vector_type(8)));
typedef float    v4f  __attribute__((ext_vector_type(4)));
typedef int      v4i  __attribute__((ext_vector_type(4)));

#define B_    8
#define T_    4096
#define F_    512
#define H_    8
#define DK_   64
#define NB_   64
#define NEGV  (-1.0e20f)
#define SCALE 0.125f

#if __has_builtin(__builtin_amdgcn_global_load_async_to_lds_b128) && \
    __has_builtin(__builtin_amdgcn_s_wait_asynccnt)
#define HAVE_ASYNC_LDS 1
#else
#define HAVE_ASYNC_LDS 0
#endif

__device__ __forceinline__ v8f wmma16(v16h a, v16h b, v8f c) {
  return __builtin_amdgcn_wmma_f32_16x16x32_f16(
      false, a, false, b, (short)0, c, false, false);
}

// A fragment 16x32 f16: lane holds row (lane&15); K = {(l>>4)*8..+7} and {16+(l>>4)*8..+7}
__device__ __forceinline__ v16h frag_a_f16(const _Float16* p, int lane) {
  const int hi = (lane >> 4) * 8;
  v8h lo = *(const v8h*)(p + hi);
  v8h hh = *(const v8h*)(p + 16 + hi);
  v16h r;
#pragma unroll
  for (int i = 0; i < 8; ++i) { r[i] = lo[i]; r[i + 8] = hh[i]; }
  return r;
}

__device__ __forceinline__ v16h frag_a_f32(const float* p, int lane) {
  const int hi = (lane >> 4) * 8;
  v4f a0 = *(const v4f*)(p + hi);
  v4f a1 = *(const v4f*)(p + hi + 4);
  v4f a2 = *(const v4f*)(p + 16 + hi);
  v4f a3 = *(const v4f*)(p + 16 + hi + 4);
  v16h r;
#pragma unroll
  for (int i = 0; i < 4; ++i) {
    r[i]      = (_Float16)a0[i];
    r[i + 4]  = (_Float16)a1[i];
    r[i + 8]  = (_Float16)a2[i];
    r[i + 12] = (_Float16)a3[i];
  }
  return r;
}

// B fragment 32x16 f16: lane holds column (lane&15); K = {(l>>4)*8..+7} and {16+(l>>4)*8..+7}
__device__ __forceinline__ v16h frag_b_f16(const _Float16* p, int lane) {
  const int hi = (lane >> 4) * 8;
  v8h lo = *(const v8h*)(p + hi);
  v8h hh = *(const v8h*)(p + 16 + hi);
  v16h r;
#pragma unroll
  for (int i = 0; i < 8; ++i) { r[i] = lo[i]; r[i + 8] = hh[i]; }
  return r;
}

// ---------------------------------------------------------------------------
// K0: weight conversion f32 -> f16.  grid = (256, 4), block 256.
// ---------------------------------------------------------------------------
__global__ __launch_bounds__(256)
void cvt_w_kernel(const float* __restrict__ Wq, const float* __restrict__ Wk,
                  const float* __restrict__ Wv, const float* __restrict__ Wo,
                  _Float16* __restrict__ w16) {
  const int m = blockIdx.y;
  const float* src = (m == 0) ? Wq : (m == 1) ? Wk : (m == 2) ? Wv : Wo;
  _Float16* dst = w16 + (size_t)m * F_ * F_;
  const int i = (blockIdx.x * 256 + threadIdx.x) * 4;
  v4f x = *(const v4f*)(src + i);
  v4h h;
#pragma unroll
  for (int j = 0; j < 4; ++j) h[j] = (_Float16)x[j];
  *(v4h*)(dst + i) = h;
}

// ---------------------------------------------------------------------------
// K1: fused QKV projection, A-resident.  grid = (512 M-tiles, 3 proj), block 128.
// ---------------------------------------------------------------------------
__global__ __launch_bounds__(128)
void qkv_proj_kernel(const float* __restrict__ xq, const float* __restrict__ xk,
                     const float* __restrict__ xv,
                     const float* __restrict__ bq, const float* __restrict__ bk,
                     const float* __restrict__ bv,
                     const _Float16* __restrict__ w16,
                     _Float16* __restrict__ q16, _Float16* __restrict__ k16,
                     _Float16* __restrict__ v16) {
  const int lane = threadIdx.x & 31;
  const int w    = threadIdx.x >> 5;
  const int proj = blockIdx.y;        // 0=q 1=k 2=v

  const float*    x    = (proj == 0) ? xq : (proj == 1) ? xk : xv;
  const float*    bias = (proj == 0) ? bq : (proj == 1) ? bk : bv;
  const _Float16* Wm   = w16 + (size_t)proj * F_ * F_;
  _Float16*       dst  = (proj == 0) ? q16 : (proj == 1) ? k16 : v16;

  const int mBase = blockIdx.x * 64 + w * 16;
  const int mRow  = mBase + (lane & 15);

  v16h a[16];                         // whole K=512 of A, resident
#pragma unroll
  for (int kk = 0; kk < 16; ++kk)
    a[kk] = frag_a_f32(x + (size_t)mRow * F_ + kk * 32, lane);

  const int rbase = mBase + ((lane & 16) >> 1);
#pragma unroll 1
  for (int nt = 0; nt < 32; ++nt) {   // 32 x 16-wide column tiles = 512 cols
    const int col = nt * 16 + (lane & 15);
    const _Float16* wp = Wm + (size_t)col * F_;
    v8f acc = {};
#pragma unroll
    for (int kk = 0; kk < 16; ++kk)
      acc = wmma16(a[kk], frag_b_f16(wp + kk * 32, lane), acc);
    const float bb = bias[nt * 16 + (lane & 15)];
#pragma unroll
    for (int j = 0; j < 8; ++j)
      dst[(size_t)(rbase + j) * F_ + col] = (_Float16)(acc[j] + bb);
  }
}

// ---------------------------------------------------------------------------
// K2: local attention.  grid = (nb=64, B*H=64), block 128 (4 waves).
// Score cols: 0..63 own block; 64..95 prev-32 keys (rows<32); 96..127 next-32.
// ---------------------------------------------------------------------------
__global__ __launch_bounds__(128)
void local_attn_kernel(const _Float16* __restrict__ q16,
                       const _Float16* __restrict__ k16,
                       const _Float16* __restrict__ v16,
                       _Float16* __restrict__ ax16) {
  __shared__ __align__(32) float    sS[64 * 128];    // raw scores f32
  __shared__ __align__(32) _Float16 sP[64 * 128];    // probs f16 (A-frag layout)
  __shared__ __align__(32) _Float16 sVraw[128 * 64]; // V rows [key'][dv]
  __shared__ __align__(32) _Float16 sVT[64 * 128];   // V^T [dv][key'] (B-frag layout)

  const int lane = threadIdx.x & 31;
  const int w    = threadIdx.x >> 5;
  const int n    = blockIdx.x;
  const int b    = blockIdx.y >> 3;
  const int h    = blockIdx.y & 7;
  const int tok0 = n * 64;

  // ---- phase 0: kick off V staging (async copy overlaps the WMMAs below) ----
#pragma unroll
  for (int it = 0; it < 8; ++it) {
    const int idx = it * 128 + threadIdx.x;  // (key col c, dv octet)
    const int c = idx >> 3;
    const int o = (idx & 7) * 8;
    int tok; bool valid = true;              // valid is workgroup-uniform per it
    if (c < 64)      { tok = tok0 + c; }
    else if (c < 96) { tok = tok0 - 32 + (c - 64); valid = (n > 0); }
    else             { tok = tok0 + 64 + (c - 96); valid = (n < NB_ - 1); }
    _Float16* lp = sVraw + c * 64 + o;
    if (valid) {
      const _Float16* gp = v16 + (size_t)(b * T_ + tok) * F_ + h * DK_ + o;
#if HAVE_ASYNC_LDS
      __builtin_amdgcn_global_load_async_to_lds_b128(
          (v4i*)(uintptr_t)gp, (v4i*)(uintptr_t)lp, 0, 0);
#else
      *(v8h*)lp = *(const v8h*)gp;
#endif
    } else {
      v8h z = {};
      *(v8h*)lp = z;
    }
  }

  // ---- S = Q K^T (own block 4 tiles + shifted half 2 tiles) ----
  const int qtok = tok0 + w * 16 + (lane & 15);
  v16h aq[2];
#pragma unroll
  for (int ks = 0; ks < 2; ++ks)
    aq[ks] = frag_a_f16(q16 + (size_t)(b * T_ + qtok) * F_ + h * DK_ + ks * 32, lane);

  v8f acc1[4] = {};
#pragma unroll
  for (int ks = 0; ks < 2; ++ks) {
#pragma unroll
    for (int nt = 0; nt < 4; ++nt) {
      const int ktok = tok0 + nt * 16 + (lane & 15);
      v16h kf = frag_b_f16(k16 + (size_t)(b * T_ + ktok) * F_ + h * DK_ + ks * 32, lane);
      acc1[nt] = wmma16(aq[ks], kf, acc1[nt]);
    }
  }

  const bool firstHalf = (w < 2);                       // wave-uniform
  const bool hasExtra  = firstHalf ? (n > 0) : (n < NB_ - 1);
  const int  etok0     = firstHalf ? (tok0 - 32) : (tok0 + 64);
  const int  ecol0     = firstHalf ? 64 : 96;
  v8f acc2[2] = {};
  if (hasExtra) {
#pragma unroll
    for (int ks = 0; ks < 2; ++ks) {
#pragma unroll
      for (int nt = 0; nt < 2; ++nt) {
        const int ktok = etok0 + nt * 16 + (lane & 15);
        v16h kf = frag_b_f16(k16 + (size_t)(b * T_ + ktok) * F_ + h * DK_ + ks * 32, lane);
        acc2[nt] = wmma16(aq[ks], kf, acc2[nt]);
      }
    }
  }

  // ---- spill scores to LDS ----
  {
    const int rbase = w * 16 + ((lane & 16) >> 1);
#pragma unroll
    for (int nt = 0; nt < 4; ++nt) {
      const int c = nt * 16 + (lane & 15);
#pragma unroll
      for (int j = 0; j < 8; ++j) sS[(rbase + j) * 128 + c] = acc1[nt][j];
    }
#pragma unroll
    for (int nt = 0; nt < 2; ++nt) {
      const int c = ecol0 + nt * 16 + (lane & 15);
#pragma unroll
      for (int j = 0; j < 8; ++j)
        sS[(rbase + j) * 128 + c] = hasExtra ? acc2[nt][j] : NEGV;
    }
  }
#if HAVE_ASYNC_LDS
  __builtin_amdgcn_s_wait_asynccnt(0);   // own wave's async V copies landed
#endif
  __syncthreads();                       // all waves' copies + scores visible

  // ---- softmax (threads 0..63) || V transpose (threads 64..127) ----
  if (threadIdx.x < 64) {
    const int r  = threadIdx.x;
    const int eb = (r < 32) ? 64 : 96;
    const int ob = (r < 32) ? 96 : 64;
    float* row = sS + r * 128;
    float mx = NEGV;
    for (int c = 0; c < 64; ++c)       mx = fmaxf(mx, row[c]);
    for (int c = eb; c < eb + 32; ++c) mx = fmaxf(mx, row[c]);
    mx *= SCALE;
    float sum = 0.f;
    for (int c = 0; c < 64; ++c)       { float e = __expf(row[c] * SCALE - mx); row[c] = e; sum += e; }
    for (int c = eb; c < eb + 32; ++c) { float e = __expf(row[c] * SCALE - mx); row[c] = e; sum += e; }
    const float inv = 1.f / sum;
    _Float16* pr = sP + r * 128;
    for (int c = 0; c < 64; ++c)       pr[c] = (_Float16)(row[c] * inv);
    for (int c = eb; c < eb + 32; ++c) pr[c] = (_Float16)(row[c] * inv);
    for (int c = ob; c < ob + 32; ++c) pr[c] = (_Float16)0.f;
  } else {
    const int t = threadIdx.x - 64;      // 0..63: transpose sVraw -> sVT
#pragma unroll 1
    for (int it = 0; it < 16; ++it) {
      const int u = it * 64 + t;         // (key col c, dv octet)
      const int c = u >> 3;
      const int o = (u & 7) * 8;
      v8h vv = *(const v8h*)(sVraw + c * 64 + o);
#pragma unroll
      for (int i = 0; i < 8; ++i) sVT[(o + i) * 128 + c] = vv[i];
    }
  }
  __syncthreads();

  // ---- O = P @ V  (K=128: 4 k-steps x 4 dv-tiles) ----
  v8f accO[4] = {};
  const int prow = w * 16 + (lane & 15);
#pragma unroll
  for (int ks = 0; ks < 4; ++ks) {
    v16h ap = frag_a_f16(sP + prow * 128 + ks * 32, lane);
#pragma unroll
    for (int nt = 0; nt < 4; ++nt) {
      v16h vf = frag_b_f16(sVT + (nt * 16 + (lane & 15)) * 128 + ks * 32, lane);
      accO[nt] = wmma16(ap, vf, accO[nt]);
    }
  }
  {
    const int rbase = w * 16 + ((lane & 16) >> 1);
#pragma unroll
    for (int nt = 0; nt < 4; ++nt) {
      const int dv = nt * 16 + (lane & 15);
#pragma unroll
      for (int j = 0; j < 8; ++j) {
        const int tokr = tok0 + rbase + j;
        ax16[(size_t)(b * T_ + tokr) * F_ + h * DK_ + dv] = (_Float16)accO[nt][j];
      }
    }
  }
}

// ---------------------------------------------------------------------------
// K3: output projection, A-resident.  grid = (512), block 128, f32 out.
// ---------------------------------------------------------------------------
__global__ __launch_bounds__(128)
void out_proj_kernel(const _Float16* __restrict__ ax16,
                     const _Float16* __restrict__ wo16,
                     const float* __restrict__ bo, float* __restrict__ out) {
  const int lane  = threadIdx.x & 31;
  const int w     = threadIdx.x >> 5;
  const int mBase = blockIdx.x * 64 + w * 16;
  const int mRow  = mBase + (lane & 15);

  v16h a[16];
#pragma unroll
  for (int kk = 0; kk < 16; ++kk)
    a[kk] = frag_a_f16(ax16 + (size_t)mRow * F_ + kk * 32, lane);

  const int rbase = mBase + ((lane & 16) >> 1);
#pragma unroll 1
  for (int nt = 0; nt < 32; ++nt) {
    const int col = nt * 16 + (lane & 15);
    const _Float16* wp = wo16 + (size_t)col * F_;
    v8f acc = {};
#pragma unroll
    for (int kk = 0; kk < 16; ++kk)
      acc = wmma16(a[kk], frag_b_f16(wp + kk * 32, lane), acc);
    const float bb = bo[nt * 16 + (lane & 15)];
#pragma unroll
    for (int j = 0; j < 8; ++j)
      out[(size_t)(rbase + j) * F_ + col] = acc[j] + bb;
  }
}

// ---------------------------------------------------------------------------
extern "C" void kernel_launch(void* const* d_in, const int* in_sizes, int n_in,
                              void* d_out, int out_size, void* d_ws, size_t ws_size,
                              hipStream_t stream) {
  const float* query  = (const float*)d_in[0];
  const float* key_in = (const float*)d_in[1];
  const float* value  = (const float*)d_in[2];
  const float* Wq = (const float*)d_in[3];  const float* bq = (const float*)d_in[4];
  const float* Wk = (const float*)d_in[5];  const float* bk = (const float*)d_in[6];
  const float* Wv = (const float*)d_in[7];  const float* bv = (const float*)d_in[8];
  const float* Wo = (const float*)d_in[9];  const float* bo = (const float*)d_in[10];
  float* out = (float*)d_out;

  const size_t ntok = (size_t)B_ * T_;      // 32768
  _Float16* q16  = (_Float16*)d_ws;
  _Float16* k16  = q16 + ntok * F_;
  _Float16* v16  = k16 + ntok * F_;
  _Float16* ax16 = v16 + ntok * F_;
  _Float16* w16  = ax16 + ntok * F_;        // 4 x 512x512 f16; total 136,314,880 B

  cvt_w_kernel<<<dim3(F_ * F_ / 1024, 4), dim3(256), 0, stream>>>(Wq, Wk, Wv, Wo, w16);
  qkv_proj_kernel<<<dim3((unsigned)(ntok / 64), 3), dim3(128), 0, stream>>>(
      query, key_in, value, bq, bk, bv, w16, q16, k16, v16);
  local_attn_kernel<<<dim3(NB_, B_ * H_), dim3(128), 0, stream>>>(q16, k16, v16, ax16);
  out_proj_kernel<<<dim3((unsigned)(ntok / 64), 1), dim3(128), 0, stream>>>(
      ax16, w16 + (size_t)3 * F_ * F_, bo, out);
}
